// LTMPLocal_73083163509153
// MI455X (gfx1250) — compile-verified, matrix operand order checked by reading, not run on hardware
//
#include <hip/hip_runtime.h>
#include <hip/hip_bf16.h>

// ---------------------------------------------------------------------------
// MI455X (gfx1250, wave32) TriangleDistToZ block.
// Bandwidth-bound (~755MB mandatory traffic vs ~175 GFLOP):
//  - all large intermediates bf16 (halves HBM traffic),
//  - all matmuls on v_wmma_f32_16x16x32_bf16 (f32 accumulate),
//  - channel-major [c][384][384] layouts make the 128 per-channel triangle
//    einsums contiguous row-major GEMMs,
//  - ALL WMMA operands staged in LDS in fragment-ready order: each fragment
//    load is two contiguous ds_load_b128,
//  - triangle GEMM double-buffers K-tiles (stage t+1 while computing t),
//  - projection kernels are branch-free inside their WMMA loops (no
//    accumulator v_mov churn from control-flow joins).
// ---------------------------------------------------------------------------

typedef __bf16 bf16_t;
typedef __bf16 v16bf __attribute__((ext_vector_type(16)));
typedef float  v8f   __attribute__((ext_vector_type(8)));

#define NDIM 384
#define NNPOS (384 * 384)
#define ECH 256
#define CCH 128

// ----- WMMA helpers (CDNA5 wave32 fragment layouts per ISA 7.12.2) ---------

__device__ __forceinline__ v8f wmma_bf16(v16bf a, v16bf b, v8f c) {
  return __builtin_amdgcn_wmma_f32_16x16x32_bf16(
      false, a, false, b, (short)0, c, false, false);
}

// Row-major fragment load: lane L<16 -> row (r0+L), K elems {0..7,16..23};
// lanes 16..31 hold K elems {8..15,24..31}. Two contiguous 16B LDS reads.
// Used for A fragments from [M][K] tiles AND B fragments from [N][K]
// (pre-transposed) tiles.
__device__ __forceinline__ v16bf load_frag_rowmajor(const bf16_t* s, int ld,
                                                    int r0, int k0, int lane) {
  const bf16_t* p = s + (size_t)(r0 + (lane & 15)) * ld + k0 + ((lane >> 4) << 3);
  v16bf f;
#pragma unroll
  for (int e = 0; e < 8; ++e) f[e] = p[e];
#pragma unroll
  for (int e = 0; e < 8; ++e) f[e + 8] = p[e + 16];
  return f;
}

__device__ __forceinline__ float sigmoidf_fast(float x) {
  return 1.0f / (1.0f + __expf(-x));
}

// Shared LN helper: normalize 64 rows of [*, 256] into Xs (bf16), one wave
// per row, 8 elems/lane, shfl_xor reduction.
#define A_XLD 264   // Xs stride over K=256 (+8 pad)

__device__ __forceinline__ void ln_rows_to_lds(const float* __restrict__ Xg,
                                               int p0,
                                               const float* __restrict__ ln_g,
                                               const float* __restrict__ ln_b,
                                               bf16_t* Xs, int lane, int wave) {
#pragma unroll
  for (int rr = 0; rr < 8; ++rr) {
    int row = wave * 8 + rr;
    const float* xr = Xg + (size_t)(p0 + row) * ECH;
    float vals[8];
    float s = 0.f, s2 = 0.f;
#pragma unroll
    for (int e = 0; e < 8; ++e) {
      float v = xr[lane + e * 32];
      vals[e] = v; s += v; s2 += v * v;
    }
#pragma unroll
    for (int off = 16; off > 0; off >>= 1) {
      s += __shfl_xor(s, off, 32);
      s2 += __shfl_xor(s2, off, 32);
    }
    float m = s * (1.f / 256.f);
    float var = s2 * (1.f / 256.f) - m * m;
    float inv = rsqrtf(var + 1e-5f);
#pragma unroll
    for (int e = 0; e < 8; ++e) {
      int col = lane + e * 32;
      Xs[(size_t)row * A_XLD + col] =
          (bf16_t)((vals[e] - m) * inv * ln_g[col] + ln_b[col]);
    }
  }
}

// ===========================================================================
// Kernel A1: fused LayerNorm + gated projections (branch-free WMMA loop).
// grid = (NNPOS/64, 6 jobs), block = 256 (8 waves).
//  job 0..2 : zn -> ab1/ab2/ab3 (masked), channel-major bf16
//  job 3    : LN(protein_pair) -> pp,   channel-major
//  job 4    : LN(l_pair) -> lp, stored TRANSPOSED (lpT[c][k][j] = lp[j][k][c])
//  job 5    : LN(inter_pair) -> ip,     channel-major
// Weights staged TRANSPOSED in LDS: W_s[n][k], stride A_WLD, so B-fragments
// are contiguous ds_load_b128 pairs.
// ===========================================================================

#define A_WLD 264   // transposed weight stride over K=256 (+8 pad)
#define A_TLD 72    // transpose-staging stride (64 + 8 pad)

__global__ __launch_bounds__(256) void ltmp_ln_gateproj(
    const float* __restrict__ z, const float* __restrict__ l_pair,
    const float* __restrict__ protein_pair, const float* __restrict__ inter_pair,
    const float* __restrict__ z_mask,
    const float* __restrict__ ln_g, const float* __restrict__ ln_b,
    const float* __restrict__ Wg1, const float* __restrict__ bg1,
    const float* __restrict__ Wg2, const float* __restrict__ bg2,
    const float* __restrict__ Wg3, const float* __restrict__ bg3,
    const float* __restrict__ W1, const float* __restrict__ b1,
    const float* __restrict__ W2, const float* __restrict__ b2,
    const float* __restrict__ W3, const float* __restrict__ b3,
    bf16_t* __restrict__ ab1c, bf16_t* __restrict__ ab2c,
    bf16_t* __restrict__ ab3c, bf16_t* __restrict__ ppc,
    bf16_t* __restrict__ lpTc, bf16_t* __restrict__ ipc) {
  const int tid = threadIdx.x;
  const int lane = tid & 31;
  const int wave = tid >> 5;
  const int job = blockIdx.y;
  const int p0 = blockIdx.x * 64;

  extern __shared__ char smem[];
  bf16_t* Wg_s = (bf16_t*)smem;                 // [128 n][A_WLD] transposed
  bf16_t* Wl_s = Wg_s + 128 * A_WLD;            // [128 n][A_WLD] transposed
  bf16_t* Xs = Wl_s + 128 * A_WLD;              // [64][A_XLD], reused as Ts
  bf16_t* Ts = Xs;                              // [128][A_TLD] after barrier

  const float* Xg = z;
  const float* WgP = Wg1; const float* bgP = bg1;
  const float* WlP = W1;  const float* blP = b1;
  bf16_t* outc = ab1c;
  bool domask = false, isLpT = false;
  switch (job) {
    case 0: Xg = z; WgP = Wg1; bgP = bg1; WlP = W1; blP = b1; domask = true; outc = ab1c; break;
    case 1: Xg = z; WgP = Wg2; bgP = bg2; WlP = W2; blP = b2; domask = true; outc = ab2c; break;
    case 2: Xg = z; WgP = Wg3; bgP = bg3; WlP = W3; blP = b3; domask = true; outc = ab3c; break;
    case 3: Xg = protein_pair; WgP = Wg2; bgP = bg2; WlP = W2; blP = b2; outc = ppc; break;
    case 4: Xg = l_pair; WgP = Wg1; bgP = bg1; WlP = W1; blP = b1; outc = lpTc; isLpT = true; break;
    default: Xg = inter_pair; WgP = Wg1; bgP = bg1; WlP = W1; blP = b1; outc = ipc; break;
  }

  // ---- Stage both weight matrices transposed (f32 -> bf16): W_s[n][k] -----
  for (int idx = tid; idx < 256 * 16; idx += 256) {  // 256 k-rows x 16 chunks
    int k = idx >> 4, nv = (idx & 15) * 8;
    const float* srcg = WgP + (size_t)k * 128 + nv;
    const float* srcl = WlP + (size_t)k * 128 + nv;
    float4 g0 = *(const float4*)(srcg);
    float4 g1 = *(const float4*)(srcg + 4);
    float4 l0 = *(const float4*)(srcl);
    float4 l1 = *(const float4*)(srcl + 4);
    Wg_s[(nv + 0) * A_WLD + k] = (bf16_t)g0.x;
    Wg_s[(nv + 1) * A_WLD + k] = (bf16_t)g0.y;
    Wg_s[(nv + 2) * A_WLD + k] = (bf16_t)g0.z;
    Wg_s[(nv + 3) * A_WLD + k] = (bf16_t)g0.w;
    Wg_s[(nv + 4) * A_WLD + k] = (bf16_t)g1.x;
    Wg_s[(nv + 5) * A_WLD + k] = (bf16_t)g1.y;
    Wg_s[(nv + 6) * A_WLD + k] = (bf16_t)g1.z;
    Wg_s[(nv + 7) * A_WLD + k] = (bf16_t)g1.w;
    Wl_s[(nv + 0) * A_WLD + k] = (bf16_t)l0.x;
    Wl_s[(nv + 1) * A_WLD + k] = (bf16_t)l0.y;
    Wl_s[(nv + 2) * A_WLD + k] = (bf16_t)l0.z;
    Wl_s[(nv + 3) * A_WLD + k] = (bf16_t)l0.w;
    Wl_s[(nv + 4) * A_WLD + k] = (bf16_t)l1.x;
    Wl_s[(nv + 5) * A_WLD + k] = (bf16_t)l1.y;
    Wl_s[(nv + 6) * A_WLD + k] = (bf16_t)l1.z;
    Wl_s[(nv + 7) * A_WLD + k] = (bf16_t)l1.w;
  }

  ln_rows_to_lds(Xg, p0, ln_g, ln_b, Xs, lane, wave);
  __syncthreads();

  // ---- Branch-free WMMA GEMM: X[64x256] @ {Wg,Wl}^T -----------------------
  const int m0 = (wave >> 1) * 16;       // 4 row groups of 16
  const int ncol0 = (wave & 1) * 64;     // 2 column halves of 64
  v8f accG[4] = {};
  v8f accP[4] = {};
#pragma unroll
  for (int kk = 0; kk < 256; kk += 32) {
    v16bf af = load_frag_rowmajor(Xs, A_XLD, m0, kk, lane);
#pragma unroll
    for (int nt = 0; nt < 4; ++nt) {
      int n0 = ncol0 + nt * 16;
      v16bf bg_f = load_frag_rowmajor(Wg_s, A_WLD, n0, kk, lane);
      v16bf bl_f = load_frag_rowmajor(Wl_s, A_WLD, n0, kk, lane);
      accG[nt] = wmma_bf16(af, bg_f, accG[nt]);
      accP[nt] = wmma_bf16(af, bl_f, accP[nt]);
    }
  }

  __syncthreads();  // all waves done reading Xs before reusing it as Ts
#pragma unroll
  for (int nt = 0; nt < 4; ++nt) {
#pragma unroll
    for (int r = 0; r < 8; ++r) {
      int rowT = m0 + r + ((lane >> 4) << 3);
      int coln = ncol0 + nt * 16 + (lane & 15);
      float gv = accG[nt][r] + bgP[coln];
      float pv = accP[nt][r] + blP[coln];
      float val = sigmoidf_fast(gv) * pv;
      if (domask) val *= z_mask[p0 + rowT];
      Ts[(size_t)coln * A_TLD + rowT] = (bf16_t)val;   // transpose stage
    }
  }
  __syncthreads();

  // ---- Channel-major writeout: thread = (channel, half of 64 positions) ---
  const int c = tid >> 1;
  const int half = tid & 1;
  if (!isLpT) {
    // 4x 16B vector copies (both sides 16B aligned)
    const uint4* src = (const uint4*)(Ts + (size_t)c * A_TLD + half * 32);
    uint4* dst = (uint4*)(outc + (size_t)c * NNPOS + p0 + half * 32);
#pragma unroll
    for (int q = 0; q < 4; ++q) dst[q] = src[q];
  } else {
    // p = j*384 + k (j fixed within the 64-tile); target lpT[c][k][j]
    int jrow = p0 / NDIM;
    int kbase = (p0 % NDIM) + half * 32;
    bf16_t* dst = outc + (size_t)c * NNPOS + jrow;
    const bf16_t* src = Ts + (size_t)c * A_TLD + half * 32;
#pragma unroll
    for (int q = 0; q < 32; ++q) dst[(size_t)(kbase + q) * NDIM] = src[q];
  }
}

// ===========================================================================
// Kernel A2: end-gate g = sigmoid(LN(z) @ Wend + bend), full EC=256 width in
// one pass (z read once). grid = NNPOS/64, block = 256 (8 waves).
// Wave tile = 16 rows x 128 cols (8 accumulators), K=256.
// ===========================================================================

__global__ __launch_bounds__(256) void ltmp_ln_endgate(
    const float* __restrict__ z,
    const float* __restrict__ ln_g, const float* __restrict__ ln_b,
    const float* __restrict__ Wend, const float* __restrict__ bend,
    bf16_t* __restrict__ gq) {
  const int tid = threadIdx.x;
  const int lane = tid & 31;
  const int wave = tid >> 5;
  const int p0 = blockIdx.x * 64;

  extern __shared__ char smem[];
  bf16_t* WT = (bf16_t*)smem;             // [256 n][A_WLD] transposed Wend
  bf16_t* Xs = WT + 256 * A_WLD;          // [64][A_XLD]

  // Wend[k][e] -> WT[e][k]
  for (int idx = tid; idx < 256 * 32; idx += 256) {  // 256 k-rows x 32 chunks
    int k = idx >> 5, nv = (idx & 31) * 8;
    const float* src = Wend + (size_t)k * ECH + nv;
    float4 w0 = *(const float4*)(src);
    float4 w1 = *(const float4*)(src + 4);
    WT[(nv + 0) * A_WLD + k] = (bf16_t)w0.x;
    WT[(nv + 1) * A_WLD + k] = (bf16_t)w0.y;
    WT[(nv + 2) * A_WLD + k] = (bf16_t)w0.z;
    WT[(nv + 3) * A_WLD + k] = (bf16_t)w0.w;
    WT[(nv + 4) * A_WLD + k] = (bf16_t)w1.x;
    WT[(nv + 5) * A_WLD + k] = (bf16_t)w1.y;
    WT[(nv + 6) * A_WLD + k] = (bf16_t)w1.z;
    WT[(nv + 7) * A_WLD + k] = (bf16_t)w1.w;
  }

  ln_rows_to_lds(z, p0, ln_g, ln_b, Xs, lane, wave);
  __syncthreads();

  const int m0 = (wave >> 1) * 16;
  const int n00 = (wave & 1) * 128;
  v8f acc[8] = {};
#pragma unroll
  for (int kk = 0; kk < 256; kk += 32) {
    v16bf af = load_frag_rowmajor(Xs, A_XLD, m0, kk, lane);
#pragma unroll
    for (int nt = 0; nt < 8; ++nt) {
      v16bf bf = load_frag_rowmajor(WT, A_WLD, n00 + nt * 16, kk, lane);
      acc[nt] = wmma_bf16(af, bf, acc[nt]);
    }
  }

#pragma unroll
  for (int nt = 0; nt < 8; ++nt) {
#pragma unroll
    for (int r = 0; r < 8; ++r) {
      int row = m0 + r + ((lane >> 4) << 3);
      int e = n00 + nt * 16 + (lane & 15);
      float gv = sigmoidf_fast(acc[nt][r] + bend[e]);
      gq[(size_t)(p0 + row) * ECH + e] = (bf16_t)gv;
    }
  }
}

// ===========================================================================
// Kernel B: triangle updates, double-buffered K loop.
//   block[c][i][j] = sum_k pp[c][i][k]*ab1[c][k][j]
//                  + sum_k ab2[c][i][k]*lpT[c][k][j]
//                  + ab3[c][i][j]*ip[c][i][j]
// grid = (9 ij-tiles of 128x128, 128 channels), block = 256 (8 waves).
// A tiles staged [M][K]; B tiles staged TRANSPOSED [N][K]. LDS = 80KB.
// ===========================================================================

#define B_LD 40   // 32 K + 8 pad

__global__ __launch_bounds__(256) void ltmp_triangle(
    const bf16_t* __restrict__ ppc, const bf16_t* __restrict__ ab1c,
    const bf16_t* __restrict__ ab2c, const bf16_t* __restrict__ lpTc,
    const bf16_t* __restrict__ ab3c, const bf16_t* __restrict__ ipc,
    float* __restrict__ blockc) {
  const int tid = threadIdx.x;
  const int lane = tid & 31;
  const int wave = tid >> 5;
  const int c = blockIdx.y;
  const int i0 = (blockIdx.x / 3) * 128;
  const int j0 = (blockIdx.x % 3) * 128;

  const bf16_t* A1 = ppc + (size_t)c * NNPOS;    // [i][k]
  const bf16_t* B1 = ab1c + (size_t)c * NNPOS;   // [k][j]
  const bf16_t* A2 = ab2c + (size_t)c * NNPOS;   // [i][k]
  const bf16_t* B2 = lpTc + (size_t)c * NNPOS;   // [k][j]

  extern __shared__ char smem[];
  bf16_t* buf = (bf16_t*)smem;
  const int TSZ = 128 * B_LD;

  auto stage = [&](int b, int kk) {
    bf16_t* A1t = buf + (size_t)b * 4 * TSZ;
    bf16_t* A2t = A1t + TSZ;
    bf16_t* B1t = A2t + TSZ;
    bf16_t* B2t = B1t + TSZ;
    for (int idx = tid; idx < 128 * 4; idx += 256) {
      int r = idx >> 2, kc = (idx & 3) * 8;
      *(uint4*)(A1t + r * B_LD + kc) =
          *(const uint4*)(A1 + (size_t)(i0 + r) * NDIM + kk + kc);
      *(uint4*)(A2t + r * B_LD + kc) =
          *(const uint4*)(A2 + (size_t)(i0 + r) * NDIM + kk + kc);
    }
    for (int idx = tid; idx < 32 * 16; idx += 256) {
      int k = idx >> 4, jv = (idx & 15) * 8;
      bf16_t t1[8], t2[8];
      *(uint4*)t1 = *(const uint4*)(B1 + (size_t)(kk + k) * NDIM + j0 + jv);
      *(uint4*)t2 = *(const uint4*)(B2 + (size_t)(kk + k) * NDIM + j0 + jv);
#pragma unroll
      for (int e = 0; e < 8; ++e) {
        B1t[(jv + e) * B_LD + k] = t1[e];
        B2t[(jv + e) * B_LD + k] = t2[e];
      }
    }
  };

  const int m0 = (wave >> 1) * 32;       // 4 row groups of 32
  const int n0 = (wave & 1) * 64;        // 2 col halves of 64
  v8f acc[2][4] = {};

  stage(0, 0);
  __syncthreads();
  const int NSTEP = NDIM / 32;           // 12
  for (int step = 0; step < NSTEP; ++step) {
    const int cur = step & 1;
    if (step + 1 < NSTEP) stage(cur ^ 1, (step + 1) * 32);  // overlap w/ math

    bf16_t* A1t = buf + (size_t)cur * 4 * TSZ;
    bf16_t* A2t = A1t + TSZ;
    bf16_t* B1t = A2t + TSZ;
    bf16_t* B2t = B1t + TSZ;

    v16bf bfr1[4], bfr2[4];
#pragma unroll
    for (int nt = 0; nt < 4; ++nt) {
      bfr1[nt] = load_frag_rowmajor(B1t, B_LD, n0 + nt * 16, 0, lane);
      bfr2[nt] = load_frag_rowmajor(B2t, B_LD, n0 + nt * 16, 0, lane);
    }
#pragma unroll
    for (int mt = 0; mt < 2; ++mt) {
      v16bf a1 = load_frag_rowmajor(A1t, B_LD, m0 + mt * 16, 0, lane);
      v16bf a2 = load_frag_rowmajor(A2t, B_LD, m0 + mt * 16, 0, lane);
#pragma unroll
      for (int nt = 0; nt < 4; ++nt) {
        acc[mt][nt] = wmma_bf16(a1, bfr1[nt], acc[mt][nt]);
        acc[mt][nt] = wmma_bf16(a2, bfr2[nt], acc[mt][nt]);
      }
    }
    __syncthreads();
  }

  // fused elementwise term + store (channel-major f32, 64B runs per 16 lanes)
#pragma unroll
  for (int mt = 0; mt < 2; ++mt) {
#pragma unroll
    for (int nt = 0; nt < 4; ++nt) {
#pragma unroll
      for (int r = 0; r < 8; ++r) {
        int gi = i0 + m0 + mt * 16 + r + ((lane >> 4) << 3);
        int gj = j0 + n0 + nt * 16 + (lane & 15);
        size_t idx = (size_t)c * NNPOS + (size_t)gi * NDIM + gj;
        blockc[idx] = acc[mt][nt][r] + (float)ab3c[idx] * (float)ipc[idx];
      }
    }
  }
}

// ===========================================================================
// Kernel C: out = g * (LN_C(block) @ Wsum + bsum) * mask
// grid = NNPOS/64 tiles of 64 positions, block = 256 (8 waves).
// Wsum staged TRANSPOSED [e=256][k=128+pad] for contiguous fragment loads.
// ===========================================================================

#define C_KLD 136   // 128 K + 8 pad
#define C_BLD 72
#define C_XLD 136

__global__ __launch_bounds__(256) void ltmp_final(
    const float* __restrict__ blockc, const bf16_t* __restrict__ gq,
    const float* __restrict__ z_mask,
    const float* __restrict__ lnc_g, const float* __restrict__ lnc_b,
    const float* __restrict__ Wsum, const float* __restrict__ bsum,
    float* __restrict__ out) {
  const int tid = threadIdx.x;
  const int lane = tid & 31;
  const int wave = tid >> 5;
  const int p0 = blockIdx.x * 64;

  extern __shared__ char smem[];
  bf16_t* WsT = (bf16_t*)smem;                      // [256 e][C_KLD]
  float* Bt = (float*)(WsT + 256 * C_KLD);          // [128 c][C_BLD]
  bf16_t* Xs = (bf16_t*)(Bt + 128 * C_BLD);         // [64 p][C_XLD]

  // Wsum[k][e] -> WsT[e][k], coalesced float4 x2 reads
  for (int idx = tid; idx < 128 * 32; idx += 256) {
    int k = idx >> 5, nv = (idx & 31) * 8;
    const float* src = Wsum + (size_t)k * ECH + nv;
    float4 w0 = *(const float4*)(src);
    float4 w1 = *(const float4*)(src + 4);
    WsT[(nv + 0) * C_KLD + k] = (bf16_t)w0.x;
    WsT[(nv + 1) * C_KLD + k] = (bf16_t)w0.y;
    WsT[(nv + 2) * C_KLD + k] = (bf16_t)w0.z;
    WsT[(nv + 3) * C_KLD + k] = (bf16_t)w0.w;
    WsT[(nv + 4) * C_KLD + k] = (bf16_t)w1.x;
    WsT[(nv + 5) * C_KLD + k] = (bf16_t)w1.y;
    WsT[(nv + 6) * C_KLD + k] = (bf16_t)w1.z;
    WsT[(nv + 7) * C_KLD + k] = (bf16_t)w1.w;
  }
  // channel-major block -> LDS (coalesced 256B runs per channel row)
  for (int idx = tid; idx < 128 * 64; idx += 256) {
    int cc = idx >> 6, pp = idx & 63;
    Bt[cc * C_BLD + pp] = blockc[(size_t)cc * NNPOS + p0 + pp];
  }
  __syncthreads();

  // LayerNorm over C=128 per position (wave32, 4 elems/lane)
#pragma unroll
  for (int rr = 0; rr < 8; ++rr) {
    int row = wave * 8 + rr;
    float vals[4];
    float s = 0.f, s2 = 0.f;
#pragma unroll
    for (int e = 0; e < 4; ++e) {
      float v = Bt[(size_t)(lane + e * 32) * C_BLD + row];
      vals[e] = v; s += v; s2 += v * v;
    }
#pragma unroll
    for (int off = 16; off > 0; off >>= 1) {
      s += __shfl_xor(s, off, 32);
      s2 += __shfl_xor(s2, off, 32);
    }
    float m = s * (1.f / 128.f);
    float var = s2 * (1.f / 128.f) - m * m;
    float inv = rsqrtf(var + 1e-5f);
#pragma unroll
    for (int e = 0; e < 4; ++e) {
      int cc = lane + e * 32;
      Xs[(size_t)row * C_XLD + cc] =
          (bf16_t)((vals[e] - m) * inv * lnc_g[cc] + lnc_b[cc]);
    }
  }
  __syncthreads();

  // X[64x128] @ Wsum[128x256] via WMMA; wave tile = 16 rows x 128 cols
  const int m0 = (wave >> 1) * 16;
  const int n00 = (wave & 1) * 128;
  v8f acc[8] = {};
#pragma unroll
  for (int kk = 0; kk < 128; kk += 32) {
    v16bf af = load_frag_rowmajor(Xs, C_XLD, m0, kk, lane);
#pragma unroll
    for (int nt = 0; nt < 8; ++nt) {
      v16bf bf = load_frag_rowmajor(WsT, C_KLD, n00 + nt * 16, kk, lane);
      acc[nt] = wmma_bf16(af, bf, acc[nt]);
    }
  }

  // fused gate * mask epilogue, coalesced f32 stores
#pragma unroll
  for (int nt = 0; nt < 8; ++nt) {
#pragma unroll
    for (int r = 0; r < 8; ++r) {
      int row = m0 + r + ((lane >> 4) << 3);
      int e = n00 + nt * 16 + (lane & 15);
      int p = p0 + row;
      float y = acc[nt][r] + bsum[e];
      out[(size_t)p * ECH + e] = (float)gq[(size_t)p * ECH + e] * y * z_mask[p];
    }
  }
}

// ===========================================================================
// Host-side launch
// ===========================================================================

extern "C" void kernel_launch(void* const* d_in, const int* in_sizes, int n_in,
                              void* d_out, int out_size, void* d_ws, size_t ws_size,
                              hipStream_t stream) {
  (void)in_sizes; (void)n_in; (void)out_size; (void)ws_size;
  const float* z            = (const float*)d_in[0];
  const float* l_pair       = (const float*)d_in[1];
  const float* protein_pair = (const float*)d_in[2];
  const float* inter_pair   = (const float*)d_in[3];
  const float* z_mask       = (const float*)d_in[4];
  const float* ln_g  = (const float*)d_in[5];
  const float* ln_b  = (const float*)d_in[6];
  const float* lnc_g = (const float*)d_in[7];
  const float* lnc_b = (const float*)d_in[8];
  const float* Wg1 = (const float*)d_in[9];  const float* bg1 = (const float*)d_in[10];
  const float* Wg2 = (const float*)d_in[11]; const float* bg2 = (const float*)d_in[12];
  const float* Wg3 = (const float*)d_in[13]; const float* bg3 = (const float*)d_in[14];
  const float* W1  = (const float*)d_in[15]; const float* b1  = (const float*)d_in[16];
  const float* W2  = (const float*)d_in[17]; const float* b2  = (const float*)d_in[18];
  const float* W3  = (const float*)d_in[19]; const float* b3  = (const float*)d_in[20];
  const float* Wend = (const float*)d_in[21]; const float* bend = (const float*)d_in[22];
  const float* Wsum = (const float*)d_in[23]; const float* bsum = (const float*)d_in[24];
  float* out = (float*)d_out;

  // Workspace (~378 MB): 6 channel-major bf16 [C][N][N], bf16 gate [N*N][EC],
  // f32 block [C][N][N].
  char* ws = (char*)d_ws;
  const size_t szCM = (size_t)NNPOS * CCH * sizeof(bf16_t);
  bf16_t* ab1c = (bf16_t*)(ws + 0 * szCM);
  bf16_t* ab2c = (bf16_t*)(ws + 1 * szCM);
  bf16_t* ab3c = (bf16_t*)(ws + 2 * szCM);
  bf16_t* ppc  = (bf16_t*)(ws + 3 * szCM);
  bf16_t* lpTc = (bf16_t*)(ws + 4 * szCM);
  bf16_t* ipc  = (bf16_t*)(ws + 5 * szCM);
  bf16_t* gq   = (bf16_t*)(ws + 6 * szCM);
  float* blockc = (float*)(ws + 6 * szCM + (size_t)NNPOS * ECH * sizeof(bf16_t));

  const size_t ldsA1 = (size_t)(2 * 128 * A_WLD + 64 * A_XLD) * sizeof(bf16_t); // 168,960
  const size_t ldsA2 = (size_t)(256 * A_WLD + 64 * A_XLD) * sizeof(bf16_t);     // 168,960
  const size_t ldsB  = (size_t)(2 * 4 * 128 * B_LD) * sizeof(bf16_t);           // 81,920
  const size_t ldsC  = (size_t)256 * C_KLD * sizeof(bf16_t) +
                       (size_t)128 * C_BLD * sizeof(float) +
                       (size_t)64 * C_XLD * sizeof(bf16_t);                     // 123,904

  (void)hipFuncSetAttribute((const void*)ltmp_ln_gateproj,
                            hipFuncAttributeMaxDynamicSharedMemorySize, (int)ldsA1);
  (void)hipFuncSetAttribute((const void*)ltmp_ln_endgate,
                            hipFuncAttributeMaxDynamicSharedMemorySize, (int)ldsA2);
  (void)hipFuncSetAttribute((const void*)ltmp_triangle,
                            hipFuncAttributeMaxDynamicSharedMemorySize, (int)ldsB);
  (void)hipFuncSetAttribute((const void*)ltmp_final,
                            hipFuncAttributeMaxDynamicSharedMemorySize, (int)ldsC);

  ltmp_ln_gateproj<<<dim3(NNPOS / 64, 6), 256, ldsA1, stream>>>(
      z, l_pair, protein_pair, inter_pair, z_mask, ln_g, ln_b,
      Wg1, bg1, Wg2, bg2, Wg3, bg3, W1, b1, W2, b2, W3, b3,
      ab1c, ab2c, ab3c, ppc, lpTc, ipc);

  ltmp_ln_endgate<<<dim3(NNPOS / 64), 256, ldsA2, stream>>>(
      z, ln_g, ln_b, Wend, bend, gq);

  ltmp_triangle<<<dim3(9, CCH), 256, ldsB, stream>>>(
      ppc, ab1c, ab2c, lpTc, ab3c, ipc, blockc);

  ltmp_final<<<dim3(NNPOS / 64), 256, ldsC, stream>>>(
      blockc, gq, z_mask, lnc_g, lnc_b, Wsum, bsum, out);
}
